// MPNTag_13030930776114
// MI455X (gfx1250) — matrix operand order repaired and checked by reading, not compiled, not run on hardware
//
#include <hip/hip_runtime.h>
#include <hip/hip_bf16.h>

#define N_NODES 10000
#define N_EDGES 320000
#define D 128
#define H 256
#define NODE_IN 256
#define EDGE_IN 17
#define STEPS 4
#define EPB 32   // edges per block in the fused edge-MLP (2 M-tiles of 16)

typedef __attribute__((ext_vector_type(16))) _Float16 v16h;
typedef __attribute__((ext_vector_type(8)))  _Float16 v8h;
typedef __attribute__((ext_vector_type(8)))  float    v8f;

__device__ __forceinline__ v8f wmma16x16x32(v16h a, v16h b, v8f c) {
  // D = A(16x32 f16) * B(32x16 f16) + C(16x16 f32)
  return __builtin_amdgcn_wmma_f32_16x16x32_f16(false, a, false, b, (short)0, c,
                                                false, false);
}

// A fragment (16x32, f16) from LDS, row-major with leading dim `lda`.
// Contiguous k order used here (matches load_b_frag): lane<16 holds row M=lane, K=0..15; lane>=16 holds K=16..31.
__device__ __forceinline__ v16h load_a_frag(const _Float16* As, int lda, int kbase, int lane) {
  const int row  = lane & 15;
  const int half = lane >> 4;
  const _Float16* ap = As + row * lda + kbase + half * 16;
  v8h lo = *(const v8h*)(ap);
  v8h hi = *(const v8h*)(ap + 8);
  v16h a;
#pragma unroll
  for (int i = 0; i < 8; ++i) { a[i] = lo[i]; a[i + 8] = hi[i]; }
  return a;
}

// B fragment (32x16, f16) from transposed weights WT[out][in] (ldb = K dimension).
// Contiguous k order used here (matches load_a_frag): lanes 0-15 hold column N=lane, K=0..15; lanes 16-31 hold K=16..31.
__device__ __forceinline__ v16h load_b_frag(const _Float16* Bt, int ldb, int col,
                                            int kbase, int lane) {
  const int n    = lane & 15;
  const int half = lane >> 4;
  return *(const v16h*)(Bt + (size_t)(col + n) * ldb + kbase + half * 16);
}

__device__ __forceinline__ void atomic_add_f32(float* p, float v) {
  __hip_atomic_fetch_add(p, v, __ATOMIC_RELAXED, __HIP_MEMORY_SCOPE_AGENT);
}

// CDNA5 async global->LDS copy (ASYNCcnt-tracked, no VGPR round trip).
// LDS byte offset = low 32 bits of the generic shared-aperture address
// (ISA: aperture check uses addr[63:32]; addr[31:0] is the LDS address).
__device__ __forceinline__ void async_ld_b128(unsigned lds_off, const void* gptr) {
  asm volatile("global_load_async_to_lds_b128 %0, %1, off"
               : : "v"(lds_off), "v"(gptr) : "memory");
}
__device__ __forceinline__ void wait_async0() {
  asm volatile("s_wait_asynccnt 0x0" : : : "memory");
}
__device__ __forceinline__ unsigned lds_off_of(const void* p) {
  return (unsigned)(uintptr_t)p;
}

// ---------------- weight conversion: f32 [Ksrc, Out] -> f16 transposed [Out, Kdim] ----
__global__ __launch_bounds__(256)
void wt_convert_kernel(_Float16* dst, const float* src, int OutDim, int Kdim, int Ksrc) {
  int i = blockIdx.x * 256 + threadIdx.x;
  if (i >= OutDim * Kdim) return;
  int o = i / Kdim, k = i - o * Kdim;
  float v = (k < Ksrc) ? src[(size_t)k * OutDim + o] : 0.0f;
  dst[i] = (_Float16)v;
}

__global__ __launch_bounds__(256)
void zero_f32_kernel(float* p, int n) {
  int i = blockIdx.x * 256 + threadIdx.x;
  if (i < n) p[i] = 0.0f;
}

// ---------------- node embedding: nf = relu(x@Wn1+bn1)@Wn2+bn2 ----------------------
__global__ __launch_bounds__(128)
void node_embed_kernel(const float* __restrict__ x,
                       const _Float16* __restrict__ Wn1t, const float* __restrict__ bn1,
                       const _Float16* __restrict__ Wn2t, const float* __restrict__ bn2,
                       _Float16* __restrict__ nf, _Float16* __restrict__ nf0) {
  __shared__ __align__(32) _Float16 sA[16 * NODE_IN];
  __shared__ __align__(32) _Float16 sH[16 * D];
  const int tid  = threadIdx.x;
  const int row0 = blockIdx.x * 16;
  for (int i = tid; i < 16 * NODE_IN; i += 128)
    sA[i] = (_Float16)x[(size_t)row0 * NODE_IN + i];
  __syncthreads();
  const int wave = tid >> 5, lane = tid & 31;
  const int n = lane & 15, rbase = (lane >> 4) * 8;
  {  // layer 1: 256 -> 128, relu
    const int col0 = wave * 32;
    v8f acc[2] = {};
#pragma unroll
    for (int kc = 0; kc < NODE_IN / 32; ++kc) {
      v16h a = load_a_frag(sA, NODE_IN, kc * 32, lane);
#pragma unroll
      for (int t = 0; t < 2; ++t)
        acc[t] = wmma16x16x32(a, load_b_frag(Wn1t, NODE_IN, col0 + t * 16, kc * 32, lane), acc[t]);
    }
#pragma unroll
    for (int t = 0; t < 2; ++t) {
      int c = col0 + t * 16 + n;
      float bv = bn1[c];
#pragma unroll
      for (int r = 0; r < 8; ++r) {
        float v = acc[t][r] + bv;
        sH[(rbase + r) * D + c] = (_Float16)(v > 0.0f ? v : 0.0f);
      }
    }
  }
  __syncthreads();
  {  // layer 2: 128 -> 128, no relu
    const int col0 = wave * 32;
    v8f acc[2] = {};
#pragma unroll
    for (int kc = 0; kc < D / 32; ++kc) {
      v16h a = load_a_frag(sH, D, kc * 32, lane);
#pragma unroll
      for (int t = 0; t < 2; ++t)
        acc[t] = wmma16x16x32(a, load_b_frag(Wn2t, D, col0 + t * 16, kc * 32, lane), acc[t]);
    }
#pragma unroll
    for (int t = 0; t < 2; ++t) {
      int c = col0 + t * 16 + n;
      float bv = bn2[c];
#pragma unroll
      for (int r = 0; r < 8; ++r) {
        _Float16 hv = (_Float16)(acc[t][r] + bv);
        size_t off = (size_t)(row0 + rbase + r) * D + c;
        nf[off] = hv; nf0[off] = hv;
      }
    }
  }
}

// ---------------- edge embedding: ef = relu(ea@We1+be1)@We2+be2 (K padded 17->32) ----
__global__ __launch_bounds__(128)
void edge_embed_kernel(const float* __restrict__ ea,
                       const _Float16* __restrict__ We1t, const float* __restrict__ be1,
                       const _Float16* __restrict__ We2t, const float* __restrict__ be2,
                       _Float16* __restrict__ ef, _Float16* __restrict__ ef0) {
  __shared__ __align__(32) _Float16 sA[16 * 32];
  __shared__ __align__(32) _Float16 sH[16 * D];
  const int tid = threadIdx.x;
  const int e0  = blockIdx.x * 16;
  for (int i = tid; i < 16 * 32; i += 128) {
    int r = i >> 5, c = i & 31;
    float v = (c < EDGE_IN) ? ea[(size_t)(e0 + r) * EDGE_IN + c] : 0.0f;
    sA[i] = (_Float16)v;
  }
  __syncthreads();
  const int wave = tid >> 5, lane = tid & 31;
  const int n = lane & 15, rbase = (lane >> 4) * 8;
  {  // layer 1: 32 -> 128, relu
    const int col0 = wave * 32;
    v8f acc[2] = {};
    v16h a = load_a_frag(sA, 32, 0, lane);
#pragma unroll
    for (int t = 0; t < 2; ++t)
      acc[t] = wmma16x16x32(a, load_b_frag(We1t, 32, col0 + t * 16, 0, lane), acc[t]);
#pragma unroll
    for (int t = 0; t < 2; ++t) {
      int c = col0 + t * 16 + n;
      float bv = be1[c];
#pragma unroll
      for (int r = 0; r < 8; ++r) {
        float v = acc[t][r] + bv;
        sH[(rbase + r) * D + c] = (_Float16)(v > 0.0f ? v : 0.0f);
      }
    }
  }
  __syncthreads();
  {  // layer 2: 128 -> 128, no relu
    const int col0 = wave * 32;
    v8f acc[2] = {};
#pragma unroll
    for (int kc = 0; kc < D / 32; ++kc) {
      v16h a = load_a_frag(sH, D, kc * 32, lane);
#pragma unroll
      for (int t = 0; t < 2; ++t)
        acc[t] = wmma16x16x32(a, load_b_frag(We2t, D, col0 + t * 16, kc * 32, lane), acc[t]);
    }
#pragma unroll
    for (int t = 0; t < 2; ++t) {
      int c = col0 + t * 16 + n;
      float bv = be2[c];
#pragma unroll
      for (int r = 0; r < 8; ++r) {
        _Float16 hv = (_Float16)(acc[t][r] + bv);
        size_t off = (size_t)(e0 + rbase + r) * D + c;
        ef[off] = hv; ef0[off] = hv;
      }
    }
  }
}

// ---------------- fused edge MLP + scatter-add ---------------------------------------
// ef = relu(relu([nf0[src]|nf[src]|nf0[dst]|nf[dst]|ef0|ef] @ Wm1 + bm1) @ Wm2 + bm2)
// agg[dst] += ef
// 32 edges per block: each B fragment is reused for two M-tiles (halves L2 weight traffic).
__global__ __launch_bounds__(128)
void edge_mlp_kernel(const _Float16* __restrict__ nf0, const _Float16* __restrict__ nf,
                     const _Float16* __restrict__ ef0, _Float16* __restrict__ ef,
                     const int* __restrict__ eidx,
                     const _Float16* __restrict__ Wm1t, const float* __restrict__ bm1,
                     const _Float16* __restrict__ Wm2t, const float* __restrict__ bm2,
                     float* __restrict__ agg) {
  __shared__ __align__(32) _Float16 sM[EPB * 6 * D];   // 32 x 768 f16 = 48KB
  __shared__ __align__(32) _Float16 sH[EPB * H];       // 32 x 256 f16 = 16KB
  __shared__ int sSrc[EPB], sDst[EPB];
  const int tid = threadIdx.x;
  const int e0  = blockIdx.x * EPB;
  if (tid < EPB) {
    sSrc[tid] = eidx[e0 + tid];
    sDst[tid] = eidx[N_EDGES + e0 + tid];
  }
  __syncthreads();
  // gather-build A tile via CDNA5 async global->LDS b128 copies (16B per chunk)
  const unsigned sM_off = lds_off_of(sM);
  for (int ch = tid; ch < EPB * 96; ch += 128) {
    int r = ch / 96, cc = ch - r * 96;
    int seg = cc >> 4, off = (cc & 15) * 8;
    const _Float16* sp;
    switch (seg) {
      case 0:  sp = nf0 + (size_t)sSrc[r] * D + off; break;
      case 1:  sp = nf  + (size_t)sSrc[r] * D + off; break;
      case 2:  sp = nf0 + (size_t)sDst[r] * D + off; break;
      case 3:  sp = nf  + (size_t)sDst[r] * D + off; break;
      case 4:  sp = ef0 + (size_t)(e0 + r) * D + off; break;
      default: sp = ef  + (size_t)(e0 + r) * D + off; break;
    }
    async_ld_b128(sM_off + (unsigned)(r * (6 * D) + cc * 8) * 2u, sp);
  }
  wait_async0();
  __syncthreads();
  const int wave = tid >> 5, lane = tid & 31;
  const int n = lane & 15, rbase = (lane >> 4) * 8;
  {  // layer 1: 768 -> 256, relu; each wave: 2 M-tiles x 4 N-tiles
    const int col0 = wave * 64;
    v8f acc0[4] = {}, acc1[4] = {};
    for (int kc = 0; kc < (6 * D) / 32; ++kc) {
      v16h a0 = load_a_frag(sM, 6 * D, kc * 32, lane);
      v16h a1 = load_a_frag(sM + 16 * (6 * D), 6 * D, kc * 32, lane);
#pragma unroll
      for (int t = 0; t < 4; ++t) {
        v16h b = load_b_frag(Wm1t, 6 * D, col0 + t * 16, kc * 32, lane);
        acc0[t] = wmma16x16x32(a0, b, acc0[t]);
        acc1[t] = wmma16x16x32(a1, b, acc1[t]);
      }
    }
#pragma unroll
    for (int t = 0; t < 4; ++t) {
      int c = col0 + t * 16 + n;
      float bv = bm1[c];
#pragma unroll
      for (int r = 0; r < 8; ++r) {
        float v0 = acc0[t][r] + bv;
        float v1 = acc1[t][r] + bv;
        sH[(rbase + r) * H + c]        = (_Float16)(v0 > 0.0f ? v0 : 0.0f);
        sH[(16 + rbase + r) * H + c]   = (_Float16)(v1 > 0.0f ? v1 : 0.0f);
      }
    }
  }
  __syncthreads();
  {  // layer 2: 256 -> 128, relu; write ef (in place) and scatter-add into agg[dst]
    const int col0 = wave * 32;
    v8f acc0[2] = {}, acc1[2] = {};
#pragma unroll
    for (int kc = 0; kc < H / 32; ++kc) {
      v16h a0 = load_a_frag(sH, H, kc * 32, lane);
      v16h a1 = load_a_frag(sH + 16 * H, H, kc * 32, lane);
#pragma unroll
      for (int t = 0; t < 2; ++t) {
        v16h b = load_b_frag(Wm2t, H, col0 + t * 16, kc * 32, lane);
        acc0[t] = wmma16x16x32(a0, b, acc0[t]);
        acc1[t] = wmma16x16x32(a1, b, acc1[t]);
      }
    }
#pragma unroll
    for (int t = 0; t < 2; ++t) {
      int c = col0 + t * 16 + n;
      float bv = bm2[c];
#pragma unroll
      for (int r = 0; r < 8; ++r) {
        int row0t = rbase + r;
        int row1t = 16 + rbase + r;
        float v0 = acc0[t][r] + bv; v0 = v0 > 0.0f ? v0 : 0.0f;
        float v1 = acc1[t][r] + bv; v1 = v1 > 0.0f ? v1 : 0.0f;
        ef[(size_t)(e0 + row0t) * D + c] = (_Float16)v0;
        ef[(size_t)(e0 + row1t) * D + c] = (_Float16)v1;
        atomic_add_f32(&agg[(size_t)sDst[row0t] * D + c], v0);
        atomic_add_f32(&agg[(size_t)sDst[row1t] * D + c], v1);
      }
    }
  }
}

// ---------------- node update: nf = relu(agg@Wu+bu) ----------------------------------
__global__ __launch_bounds__(128)
void node_update_kernel(const float* __restrict__ agg,
                        const _Float16* __restrict__ Wut, const float* __restrict__ bu,
                        _Float16* __restrict__ nf) {
  __shared__ __align__(32) _Float16 sA[16 * D];
  const int tid  = threadIdx.x;
  const int row0 = blockIdx.x * 16;
  for (int i = tid; i < 16 * D; i += 128)
    sA[i] = (_Float16)agg[(size_t)row0 * D + i];
  __syncthreads();
  const int wave = tid >> 5, lane = tid & 31;
  const int n = lane & 15, rbase = (lane >> 4) * 8;
  const int col0 = wave * 32;
  v8f acc[2] = {};
#pragma unroll
  for (int kc = 0; kc < D / 32; ++kc) {
    v16h a = load_a_frag(sA, D, kc * 32, lane);
#pragma unroll
    for (int t = 0; t < 2; ++t)
      acc[t] = wmma16x16x32(a, load_b_frag(Wut, D, col0 + t * 16, kc * 32, lane), acc[t]);
  }
#pragma unroll
  for (int t = 0; t < 2; ++t) {
    int c = col0 + t * 16 + n;
    float bv = bu[c];
#pragma unroll
    for (int r = 0; r < 8; ++r) {
      float v = acc[t][r] + bv;
      nf[(size_t)(row0 + rbase + r) * D + c] = (_Float16)(v > 0.0f ? v : 0.0f);
    }
  }
}

// ---------------- tags: (relu(nf@Wt1+bt1)@Wt2+bt2).squeeze ---------------------------
__global__ __launch_bounds__(128)
void tag_kernel(const _Float16* __restrict__ nf,
                const _Float16* __restrict__ Wt1t, const float* __restrict__ bt1,
                const float* __restrict__ Wt2, const float* __restrict__ bt2,
                float* __restrict__ out) {
  __shared__ __align__(32) _Float16 sA[16 * D];
  __shared__ __align__(32) _Float16 sH[16 * D];
  const int tid  = threadIdx.x;
  const int row0 = blockIdx.x * 16;
  const unsigned sA_off = lds_off_of(sA);
  for (int i = tid; i < (16 * D) / 8; i += 128)
    async_ld_b128(sA_off + (unsigned)i * 16u, nf + (size_t)row0 * D + i * 8);
  wait_async0();
  __syncthreads();
  const int wave = tid >> 5, lane = tid & 31;
  const int n = lane & 15, rbase = (lane >> 4) * 8;
  {  // layer 1: 128 -> 128, relu
    const int col0 = wave * 32;
    v8f acc[2] = {};
#pragma unroll
    for (int kc = 0; kc < D / 32; ++kc) {
      v16h a = load_a_frag(sA, D, kc * 32, lane);
#pragma unroll
      for (int t = 0; t < 2; ++t)
        acc[t] = wmma16x16x32(a, load_b_frag(Wt1t, D, col0 + t * 16, kc * 32, lane), acc[t]);
    }
#pragma unroll
    for (int t = 0; t < 2; ++t) {
      int c = col0 + t * 16 + n;
      float bv = bt1[c];
#pragma unroll
      for (int r = 0; r < 8; ++r) {
        float v = acc[t][r] + bv;
        sH[(rbase + r) * D + c] = (_Float16)(v > 0.0f ? v : 0.0f);
      }
    }
  }
  __syncthreads();
  // layer 2: 128 -> 1, plain f32 dot per row
  if (tid < 16) {
    float s = bt2[0];
    for (int k = 0; k < D; ++k) s += (float)sH[tid * D + k] * Wt2[k];
    out[row0 + tid] = s;
  }
}

extern "C" void kernel_launch(void* const* d_in, const int* in_sizes, int n_in,
                              void* d_out, int out_size, void* d_ws, size_t ws_size,
                              hipStream_t stream) {
  const float* x    = (const float*)d_in[0];
  const float* ea   = (const float*)d_in[1];
  const int*   eidx = (const int*)d_in[2];
  const float* Wn1  = (const float*)d_in[3];  const float* bn1 = (const float*)d_in[4];
  const float* Wn2  = (const float*)d_in[5];  const float* bn2 = (const float*)d_in[6];
  const float* We1  = (const float*)d_in[7];  const float* be1 = (const float*)d_in[8];
  const float* We2  = (const float*)d_in[9];  const float* be2 = (const float*)d_in[10];
  const float* Wm1  = (const float*)d_in[11]; const float* bm1 = (const float*)d_in[12];
  const float* Wm2  = (const float*)d_in[13]; const float* bm2 = (const float*)d_in[14];
  const float* Wu   = (const float*)d_in[15]; const float* bu  = (const float*)d_in[16];
  const float* Wt1  = (const float*)d_in[17]; const float* bt1 = (const float*)d_in[18];
  const float* Wt2  = (const float*)d_in[19]; const float* bt2 = (const float*)d_in[20];
  float* out = (float*)d_out;

  char* base = (char*)d_ws;
  auto carve = [&](size_t bytes) -> void* {
    void* p = (void*)base;
    base += (bytes + 255) & ~(size_t)255;
    return p;
  };
  _Float16* nf   = (_Float16*)carve((size_t)N_NODES * D * 2);
  _Float16* nf0  = (_Float16*)carve((size_t)N_NODES * D * 2);
  _Float16* ef   = (_Float16*)carve((size_t)N_EDGES * D * 2);
  _Float16* ef0  = (_Float16*)carve((size_t)N_EDGES * D * 2);
  float*    agg  = (float*)   carve((size_t)N_NODES * D * 4);
  _Float16* Wn1t = (_Float16*)carve((size_t)D * NODE_IN * 2);
  _Float16* Wn2t = (_Float16*)carve((size_t)D * D * 2);
  _Float16* We1t = (_Float16*)carve((size_t)D * 32 * 2);
  _Float16* We2t = (_Float16*)carve((size_t)D * D * 2);
  _Float16* Wm1t = (_Float16*)carve((size_t)H * 6 * D * 2);
  _Float16* Wm2t = (_Float16*)carve((size_t)D * H * 2);
  _Float16* Wut  = (_Float16*)carve((size_t)D * D * 2);
  _Float16* Wt1t = (_Float16*)carve((size_t)D * D * 2);

  auto cv = [&](_Float16* dst, const float* src, int OutDim, int Kdim, int Ksrc) {
    int tot = OutDim * Kdim;
    wt_convert_kernel<<<(tot + 255) / 256, 256, 0, stream>>>(dst, src, OutDim, Kdim, Ksrc);
  };
  cv(Wn1t, Wn1, D, NODE_IN, NODE_IN);
  cv(Wn2t, Wn2, D, D, D);
  cv(We1t, We1, D, 32, EDGE_IN);
  cv(We2t, We2, D, D, D);
  cv(Wm1t, Wm1, H, 6 * D, 6 * D);
  cv(Wm2t, Wm2, D, H, H);
  cv(Wut,  Wu,  D, D, D);
  cv(Wt1t, Wt1, D, D, D);

  node_embed_kernel<<<N_NODES / 16, 128, 0, stream>>>(x, Wn1t, bn1, Wn2t, bn2, nf, nf0);
  edge_embed_kernel<<<N_EDGES / 16, 128, 0, stream>>>(ea, We1t, be1, We2t, be2, ef, ef0);

  for (int s = 0; s < STEPS; ++s) {
    zero_f32_kernel<<<(N_NODES * D + 255) / 256, 256, 0, stream>>>(agg, N_NODES * D);
    edge_mlp_kernel<<<N_EDGES / EPB, 128, 0, stream>>>(nf0, nf, ef0, ef, eidx,
                                                       Wm1t, bm1, Wm2t, bm2, agg);
    node_update_kernel<<<N_NODES / 16, 128, 0, stream>>>(agg, Wut, bu, nf);
  }
  tag_kernel<<<N_NODES / 16, 128, 0, stream>>>(nf, Wt1t, bt1, Wt2, bt2, out);
}
